// DeltaNetLayer_8787503087818
// MI455X (gfx1250) — matrix operand order repaired and checked by reading, not compiled, not run on hardware
//
#include <hip/hip_runtime.h>
#include <hip/hip_bf16.h>
#include <math.h>

typedef __attribute__((ext_vector_type(16))) __bf16 v16bf;
typedef __attribute__((ext_vector_type(8)))  float  v8f;
typedef unsigned short ushort_t;
typedef unsigned int   uint_t;

#define DEV __device__ __forceinline__

// ---------------- problem constants ----------------
constexpr int BB = 8, LL = 2048, DD = 1024;
constexpr int NTOK = BB * LL;        // 16384 tokens
constexpr int CHUNK = 32;            // timesteps per chunk == WMMA K dim
constexpr int NCHUNK = LL / CHUNK;   // 64
constexpr int DV = 32;               // value-dim slice per workgroup
constexpr int NSLICE = DD / DV;      // 32

// ---------------- scan-kernel LDS layout (bytes) ----------------
constexpr int SSTR  = 33;    // S row stride (floats), padded vs 64 banks
constexpr int MSTR  = 33;    // 32x32 fp32 scratch row stride
constexpr int HSTR  = 34;    // bf16 tile row stride (halves)
constexpr int SBSTR = 1026;  // bf16 S-mirror row stride (halves)

constexpr size_t OFF_S    = 0;                                    // float [1024][33]
constexpr size_t OFF_KK   = OFF_S    + (size_t)DD * SSTR * 4;
constexpr size_t OFF_QK   = OFF_KK   + 32 * MSTR * 4;
constexpr size_t OFF_KS   = OFF_QK   + 32 * MSTR * 4;
constexpr size_t OFF_QS   = OFF_KS   + 32 * MSTR * 4;
constexpr size_t OFF_T    = OFF_QS   + 32 * MSTR * 4;
constexpr size_t OFF_U    = OFF_T    + 32 * MSTR * 4;
constexpr size_t OFF_VADJ = OFF_U    + 32 * MSTR * 4;
constexpr size_t OFF_VC   = OFF_VADJ + 32 * MSTR * 4;
constexpr size_t OFF_BETA = OFF_VC   + 32 * MSTR * 4;
constexpr size_t OFF_SB   = OFF_BETA + 32 * 4;                    // ushort[32][1026]
constexpr size_t OFF_KCT  = OFF_SB   + (size_t)32 * SBSTR * 2;    // ushort[1024][34]
constexpr size_t OFF_VAT  = OFF_KCT  + (size_t)DD * HSTR * 2;     // ushort[32][34]
constexpr size_t SMEM_SCAN = OFF_VAT + 32 * HSTR * 2;             // ~306,560 B < 320KB

// ---------------- small helpers ----------------
DEV ushort_t f2bf(float f) {                       // fp32 -> bf16 (RNE)
    uint_t u = __float_as_uint(f);
    u += 0x7FFFu + ((u >> 16) & 1u);
    return (ushort_t)(u >> 16);
}
DEV float bf2f(ushort_t h) { return __uint_as_float(((uint_t)h) << 16); }
DEV uint_t pack2bf(float a, float b) {
    return (uint_t)f2bf(a) | ((uint_t)f2bf(b) << 16);
}

struct FragBF { union { v16bf v; uint_t u[8]; }; };

// ---- LDS fragment loaders (4B-aligned rows, stride 34 halves) ----
// A-matrix 16x32 bf16: lane holds M=lane&15; two 8-half runs at
// K = base..base+7 and base+16..base+23, base = (lane>=16)?8:0.
DEV FragBF loadA16(const ushort_t* rowbase, int kbase, int lane) {
    FragBF f;
    const uint_t* p = (const uint_t*)(rowbase + kbase + ((lane >> 4) << 3));
#pragma unroll
    for (int i = 0; i < 4; ++i) { f.u[i] = p[i]; f.u[4 + i] = p[8 + i]; }
    return f;
}
// B-matrix 32x16 bf16: lane holds N=lane&15; two 8-half runs at
// K = base..base+7 and base+16..base+23, base = (lane>=16)?8:0.
DEV FragBF loadB16(const ushort_t* rowbase, int kbase, int lane) {
    FragBF f;
    const uint_t* p = (const uint_t*)(rowbase + kbase + ((lane >> 4) << 3));
#pragma unroll
    for (int i = 0; i < 4; ++i) { f.u[i] = p[i]; f.u[4 + i] = p[8 + i]; }
    return f;
}
// ---- global fragment loaders: rows are 2048B-aligned, runs 16B-aligned,
// so these lower to global_load_b128.
DEV FragBF loadA16_g(const ushort_t* rowbase, int kbase, int lane) {
    FragBF f;
    const uint4* p = (const uint4*)(rowbase + kbase + ((lane >> 4) << 3));
    uint4 a = p[0], b = p[2];     // second run is +16 halves = +2 uint4
    f.u[0] = a.x; f.u[1] = a.y; f.u[2] = a.z; f.u[3] = a.w;
    f.u[4] = b.x; f.u[5] = b.y; f.u[6] = b.z; f.u[7] = b.w;
    return f;
}
DEV FragBF loadB16_g(const ushort_t* rowbase, int kbase, int lane) {
    FragBF f;
    const uint4* p = (const uint4*)(rowbase + kbase + ((lane >> 4) << 3));
    uint4 a = p[0], b = p[2];     // second run is +16 halves = +2 uint4
    f.u[0] = a.x; f.u[1] = a.y; f.u[2] = a.z; f.u[3] = a.w;
    f.u[4] = b.x; f.u[5] = b.y; f.u[6] = b.z; f.u[7] = b.w;
    return f;
}
DEV v8f wmma_bf16(const FragBF& a, const FragBF& b, v8f c) {
    return __builtin_amdgcn_wmma_f32_16x16x32_bf16(
        /*neg_a=*/false, a.v, /*neg_b=*/false, b.v,
        /*c_mod=*/(short)0, c, /*reuse_a=*/false, /*reuse_b=*/false);
}

// =====================================================================
// Kernel 1: projection GEMM  Out[n][j] = epilogue( sum_d X[n][d]*W[d][j] )
// mode 0: q (scale 1/32) ; mode 1: k raw ; mode 2: v (SiLU). Output bf16.
// Block tile 128x64, 8 wave32 waves, bf16 WMMA with fp32 accumulation.
// =====================================================================
__global__ __launch_bounds__(256) void proj_gemm_kernel(
    const float* __restrict__ X, const float* __restrict__ W,
    ushort_t* __restrict__ Out, int mode) {
    __shared__ ushort_t At[128 * HSTR];   // x tile, bf16, [row][dk]
    __shared__ ushort_t Wt[64 * HSTR];    // W tile transposed, bf16, [j][dk]

    const int tid  = threadIdx.x;
    const int wave = tid >> 5, lane = tid & 31;
    const int mtile = blockIdx.x >> 4;     // 128 M-tiles
    const int ntile = blockIdx.x & 15;     // 16 N-tiles
    const int row0 = mtile * 128, col0 = ntile * 64;

    v8f acc[4];
#pragma unroll
    for (int i = 0; i < 4; ++i) acc[i] = v8f{};

    for (int kb = 0; kb < 32; ++kb) {
        const int kc = kb * 32;
        // stage X tile [128][32] (float4 loads -> packed bf16 pair stores)
        for (int idx = tid; idx < 128 * 8; idx += 256) {
            int r = idx >> 3, d4 = (idx & 7) * 4;
            float4 xv = *(const float4*)&X[(size_t)(row0 + r) * DD + kc + d4];
            *(uint_t*)&At[r * HSTR + d4]     = pack2bf(xv.x, xv.y);
            *(uint_t*)&At[r * HSTR + d4 + 2] = pack2bf(xv.z, xv.w);
        }
        // stage W tile transposed [64 j][32 dk] (float4 loads, b16 stores)
        for (int idx = tid; idx < 32 * 16; idx += 256) {
            int dk = idx >> 4, j4 = (idx & 15) * 4;
            float4 wv = *(const float4*)&W[(size_t)(kc + dk) * DD + col0 + j4];
            Wt[(j4 + 0) * HSTR + dk] = f2bf(wv.x);
            Wt[(j4 + 1) * HSTR + dk] = f2bf(wv.y);
            Wt[(j4 + 2) * HSTR + dk] = f2bf(wv.z);
            Wt[(j4 + 3) * HSTR + dk] = f2bf(wv.w);
        }
        if (kb + 1 < 32) {  // warm L2 for next K tiles (global_prefetch_b8)
            __builtin_prefetch(X + (size_t)(row0 + (tid & 127)) * DD + kc + 32, 0, 1);
            __builtin_prefetch(W + (size_t)(kc + 32 + (tid & 31)) * DD + col0, 0, 1);
        }
        __syncthreads();

        FragBF a = loadA16(At + (wave * 16 + (lane & 15)) * HSTR, 0, lane);
#pragma unroll
        for (int nt = 0; nt < 4; ++nt) {
            FragBF b = loadB16(Wt + (nt * 16 + (lane & 15)) * HSTR, 0, lane);
            acc[nt] = wmma_bf16(a, b, acc[nt]);
        }
        __syncthreads();
    }

    // epilogue + store bf16
#pragma unroll
    for (int nt = 0; nt < 4; ++nt) {
#pragma unroll
        for (int r = 0; r < 8; ++r) {
            int m = r + ((lane >> 4) << 3);                // C/D layout row
            int grow = row0 + wave * 16 + m;
            int gcol = col0 + nt * 16 + (lane & 15);
            float vv = acc[nt][r];
            if (mode == 0) vv *= 0.03125f;                 // D^-0.5
            else if (mode == 2) vv = vv / (1.0f + expf(-vv)); // SiLU
            Out[(size_t)grow * DD + gcol] = f2bf(vv);
        }
    }
}

// =====================================================================
// Kernel 2: beta[t] = sigmoid( x[t] . Wb ) ; one wave32 per token
// =====================================================================
__global__ __launch_bounds__(256) void beta_kernel(
    const float* __restrict__ X, const float* __restrict__ Wb,
    float* __restrict__ beta) {
    const int lane = threadIdx.x & 31;
    const int t = blockIdx.x * 8 + (threadIdx.x >> 5);
    float s = 0.f;
#pragma unroll
    for (int i = 0; i < 8; ++i) {
        int e = (i * 32 + lane) * 4;
        float4 xv = *(const float4*)&X[(size_t)t * DD + e];
        float4 wv = *(const float4*)&Wb[e];
        s += xv.x * wv.x + xv.y * wv.y + xv.z * wv.z + xv.w * wv.w;
    }
#pragma unroll
    for (int m = 16; m; m >>= 1) s += __shfl_xor(s, m, 32);
    if (lane == 0) beta[t] = 1.0f / (1.0f + expf(-s));
}

// =====================================================================
// Kernel 3: L2-normalize each k row in place (bf16); one wave per token
// =====================================================================
__global__ __launch_bounds__(256) void knorm_kernel(ushort_t* __restrict__ K) {
    const int lane = threadIdx.x & 31;
    const int t = blockIdx.x * 8 + (threadIdx.x >> 5);
    uint4 kv[4];
    float ss = 0.f;
#pragma unroll
    for (int i = 0; i < 4; ++i) {
        int e = (i * 32 + lane) * 8;
        kv[i] = *(const uint4*)&K[(size_t)t * DD + e];
        const uint_t* w = (const uint_t*)&kv[i];
#pragma unroll
        for (int j = 0; j < 4; ++j) {
            float lo = bf2f((ushort_t)(w[j] & 0xFFFF)), hi = bf2f((ushort_t)(w[j] >> 16));
            ss += lo * lo + hi * hi;
        }
    }
#pragma unroll
    for (int m = 16; m; m >>= 1) ss += __shfl_xor(ss, m, 32);
    const float sc = 1.0f / fmaxf(sqrtf(ss), 1e-12f);
#pragma unroll
    for (int i = 0; i < 4; ++i) {
        int e = (i * 32 + lane) * 8;
        uint_t* w = (uint_t*)&kv[i];
        uint4 outp;
        uint_t* o = (uint_t*)&outp;
#pragma unroll
        for (int j = 0; j < 4; ++j)
            o[j] = pack2bf(bf2f((ushort_t)(w[j] & 0xFFFF)) * sc,
                           bf2f((ushort_t)(w[j] >> 16)) * sc);
        *(uint4*)&K[(size_t)t * DD + e] = outp;
    }
}

// =====================================================================
// Kernel 4: chunked delta-rule scan.
// Block (b, slice): owns S[1024][32] fp32 in LDS; processes 64 chunks of 32
// steps. Per chunk (KK/QK/KS/QS and S update = v_wmma_f32_16x16x32_bf16, fp32 accum):
//   KK = K Kt, QK = Q Kt, KS = K S0, QS = Q S0
//   A  = strict_tril(diag(beta) KK);  T = (I+A)^-1  (forward subst)
//   Vadj = T (diag(beta)(V - KS))
//   O = QS + tril(QK,0) Vadj  -> global
//   S += Kt_chunk Vadj  (WMMA, C-operand accumulate in LDS fp32)
// =====================================================================
__global__ __launch_bounds__(256) void scan_kernel(
    const ushort_t* __restrict__ qg, const ushort_t* __restrict__ kg,
    const ushort_t* __restrict__ vg, const float* __restrict__ betag,
    float* __restrict__ og) {
    extern __shared__ char smem[];
    float*    S     = (float*)(smem + OFF_S);
    float*    KK    = (float*)(smem + OFF_KK);
    float*    QK    = (float*)(smem + OFF_QK);
    float*    KS    = (float*)(smem + OFF_KS);
    float*    QS    = (float*)(smem + OFF_QS);
    float*    Tm    = (float*)(smem + OFF_T);
    float*    Uv    = (float*)(smem + OFF_U);
    float*    Vadj  = (float*)(smem + OFF_VADJ);
    float*    Vc    = (float*)(smem + OFF_VC);
    float*    betaC = (float*)(smem + OFF_BETA);
    ushort_t* Sb    = (ushort_t*)(smem + OFF_SB);    // S mirror bf16 [vcol][dk]
    ushort_t* KcT   = (ushort_t*)(smem + OFF_KCT);   // K chunk bf16  [dk][t]
    ushort_t* VaT   = (ushort_t*)(smem + OFF_VAT);   // Vadj bf16     [vcol][t]

    const int tid  = threadIdx.x;
    const int wave = tid >> 5, lane = tid & 31;
    const int b = blockIdx.x >> 5;            // batch
    const int sl = blockIdx.x & 31;           // value slice
    const int c0 = sl * DV;                   // value column base

    // zero state
    for (int i = tid; i < DD * SSTR; i += 256) S[i] = 0.f;
    for (int i = tid; i < 32 * SBSTR; i += 256) Sb[i] = 0;
    __syncthreads();

    for (int ch = 0; ch < NCHUNK; ++ch) {
        const int r0 = b * LL + ch * CHUNK;   // first token of chunk

        // ---- P0: stage transposed K chunk (b128 loads), V slice, beta ----
        for (int idx = tid; idx < CHUNK * (DD / 8); idx += 256) {
            int t = idx >> 7, d = (idx & 127) * 8;
            uint4 kv = *(const uint4*)&kg[(size_t)(r0 + t) * DD + d];
            const uint_t* w = (const uint_t*)&kv;
#pragma unroll
            for (int j = 0; j < 4; ++j) {
                KcT[(d + 2 * j + 0) * HSTR + t] = (ushort_t)(w[j] & 0xFFFF);
                KcT[(d + 2 * j + 1) * HSTR + t] = (ushort_t)(w[j] >> 16);
            }
        }
        for (int idx = tid; idx < CHUNK * (DV / 2); idx += 256) {
            int t = idx >> 4, n2 = (idx & 15) * 2;
            uint_t pv = *(const uint_t*)&vg[(size_t)(r0 + t) * DD + c0 + n2];
            Vc[t * MSTR + n2]     = bf2f((ushort_t)(pv & 0xFFFF));
            Vc[t * MSTR + n2 + 1] = bf2f((ushort_t)(pv >> 16));
        }
        if (tid < 32) betaC[tid] = betag[r0 + tid];
        if (ch + 1 < NCHUNK) {
            __builtin_prefetch(kg + (size_t)(r0 + CHUNK + (tid & 31)) * DD, 0, 1);
            __builtin_prefetch(qg + (size_t)(r0 + CHUNK + (tid & 31)) * DD, 0, 1);
        }
        __syncthreads();

        // ---- P1: 16 WMMA tile jobs over 8 waves ----
        for (int job = wave; job < 16; job += 8) {
            v8f acc = v8f{};
            const int t2 = job & 3, ti = t2 >> 1, tj = t2 & 1;
            if (job < 8) {  // KK (0-3) / QK (4-7): [32 t][32 t] over dk=1024
                const bool isQ = job >= 4;
                const ushort_t* Arow =
                    (isQ ? qg : kg) + (size_t)(r0 + ti * 16 + (lane & 15)) * DD;
                const ushort_t* Brow = kg + (size_t)(r0 + tj * 16 + (lane & 15)) * DD;
                for (int kb = 0; kb < 32; ++kb) {
                    FragBF a = loadA16_g(Arow, kb * 32, lane);
                    FragBF bb = loadB16_g(Brow, kb * 32, lane);
                    acc = wmma_bf16(a, bb, acc);
                }
                float* dst = isQ ? QK : KK;
#pragma unroll
                for (int r = 0; r < 8; ++r)
                    dst[(ti * 16 + r + ((lane >> 4) << 3)) * MSTR + tj * 16 + (lane & 15)] = acc[r];
            } else {        // KS (8-11) / QS (12-15): [32 t][32 v] over dk
                const bool isQ = (job - 8) >= 4;
                const ushort_t* Arow =
                    (isQ ? qg : kg) + (size_t)(r0 + ti * 16 + (lane & 15)) * DD;
                const ushort_t* Brow = Sb + (size_t)(tj * 16 + (lane & 15)) * SBSTR;
                for (int kb = 0; kb < 32; ++kb) {
                    FragBF a = loadA16_g(Arow, kb * 32, lane);
                    FragBF bb = loadB16(Brow, kb * 32, lane);
                    acc = wmma_bf16(a, bb, acc);
                }
                float* dst = isQ ? QS : KS;
#pragma unroll
                for (int r = 0; r < 8; ++r)
                    dst[(ti * 16 + r + ((lane >> 4) << 3)) * MSTR + tj * 16 + (lane & 15)] = acc[r];
            }
        }
        __syncthreads();

        // ---- P2a: A = strict_tril(diag(beta) * KK), in place ----
        for (int idx = tid; idx < 32 * 32; idx += 256) {
            int i = idx >> 5, j = idx & 31;
            KK[i * MSTR + j] = (j < i) ? betaC[i] * KK[i * MSTR + j] : 0.f;
        }
        __syncthreads();

        // ---- P2b: T = (I+A)^-1 (wave 0, lane=column) ; others: U ----
        if (wave == 0) {
            const int c = lane;
            for (int i = 0; i < 32; ++i) {
                float sv = (i == c) ? 1.f : 0.f;
                for (int j = 0; j < i; ++j) sv -= KK[i * MSTR + j] * Tm[j * MSTR + c];
                Tm[i * MSTR + c] = sv;
            }
        } else {
            for (int idx = tid - 32; idx < 32 * 32; idx += 224) {
                int i = idx >> 5, n = idx & 31;
                Uv[i * MSTR + n] = betaC[i] * (Vc[i * MSTR + n] - KS[i * MSTR + n]);
            }
        }
        __syncthreads();

        // ---- P3: Vadj = T @ U (+ bf16 transposed copy) ----
        for (int idx = tid; idx < 32 * 32; idx += 256) {
            int i = idx >> 5, n = idx & 31;
            float sv = 0.f;
#pragma unroll
            for (int j = 0; j < 32; ++j) sv += Tm[i * MSTR + j] * Uv[j * MSTR + n];
            Vadj[i * MSTR + n] = sv;
            VaT[n * HSTR + i] = f2bf(sv);
        }
        __syncthreads();

        // ---- P4a: O = QS + tril(QK,0) @ Vadj -> global ----
        for (int idx = tid; idx < 32 * 32; idx += 256) {
            int i = idx >> 5, n = idx & 31;
            float sv = QS[i * MSTR + n];
            for (int j = 0; j <= i; ++j) sv += QK[i * MSTR + j] * Vadj[j * MSTR + n];
            og[(size_t)(r0 + i) * DD + c0 + n] = sv;
        }
        // ---- P4b: S += KcT @ Vadj (WMMA with fp32 C accumulate in LDS) ----
        for (int u = 0; u < 16; ++u) {
            const int tt = wave * 16 + u;
            const int dkb = tt >> 1, tn = tt & 1;
            FragBF a = loadA16(KcT + (size_t)(dkb * 16 + (lane & 15)) * HSTR, 0, lane);
            FragBF bb = loadB16(VaT + (size_t)(tn * 16 + (lane & 15)) * HSTR, 0, lane);
            v8f cfr;
#pragma unroll
            for (int r = 0; r < 8; ++r)
                cfr[r] = S[(dkb * 16 + r + ((lane >> 4) << 3)) * SSTR + tn * 16 + (lane & 15)];
            cfr = wmma_bf16(a, bb, cfr);
#pragma unroll
            for (int r = 0; r < 8; ++r)
                S[(dkb * 16 + r + ((lane >> 4) << 3)) * SSTR + tn * 16 + (lane & 15)] = cfr[r];
        }
        __syncthreads();

        // ---- P5: refresh bf16 mirror Sb[vcol][dk] ----
        for (int idx = tid; idx < DD * DV; idx += 256) {
            int dk = idx >> 5, n = idx & 31;
            Sb[n * SBSTR + dk] = f2bf(S[dk * SSTR + n]);
        }
        __syncthreads();
    }
}

// =====================================================================
// Kernel 5: LayerNorm over last dim; one wave32 per token (float4 I/O)
// =====================================================================
__global__ __launch_bounds__(256) void ln_kernel(
    const float* __restrict__ O, const float* __restrict__ G,
    const float* __restrict__ Bv, float* __restrict__ out) {
    const int lane = threadIdx.x & 31;
    const int t = blockIdx.x * 8 + (threadIdx.x >> 5);
    float4 v[8];
    float s = 0.f;
#pragma unroll
    for (int i = 0; i < 8; ++i) {
        v[i] = *(const float4*)&O[(size_t)t * DD + (i * 32 + lane) * 4];
        s += v[i].x + v[i].y + v[i].z + v[i].w;
    }
#pragma unroll
    for (int m = 16; m; m >>= 1) s += __shfl_xor(s, m, 32);
    const float mu = s * (1.0f / DD);
    float ss = 0.f;
#pragma unroll
    for (int i = 0; i < 8; ++i) {
        float dx = v[i].x - mu, dy = v[i].y - mu, dz = v[i].z - mu, dw = v[i].w - mu;
        ss += dx * dx + dy * dy + dz * dz + dw * dw;
    }
#pragma unroll
    for (int m = 16; m; m >>= 1) ss += __shfl_xor(ss, m, 32);
    const float rs = rsqrtf(ss * (1.0f / DD) + 1e-5f);
#pragma unroll
    for (int i = 0; i < 8; ++i) {
        int e = (i * 32 + lane) * 4;
        float4 g = *(const float4*)&G[e];
        float4 bb = *(const float4*)&Bv[e];
        float4 r;
        r.x = (v[i].x - mu) * rs * g.x + bb.x;
        r.y = (v[i].y - mu) * rs * g.y + bb.y;
        r.z = (v[i].z - mu) * rs * g.z + bb.z;
        r.w = (v[i].w - mu) * rs * g.w + bb.w;
        *(float4*)&out[(size_t)t * DD + e] = r;
    }
}

// =====================================================================
extern "C" void kernel_launch(void* const* d_in, const int* in_sizes, int n_in,
                              void* d_out, int out_size, void* d_ws, size_t ws_size,
                              hipStream_t stream) {
    const float* x    = (const float*)d_in[0];
    const float* Wq   = (const float*)d_in[1];
    const float* Wk   = (const float*)d_in[2];
    const float* Wv   = (const float*)d_in[3];
    const float* Wb   = (const float*)d_in[4];
    const float* ln_g = (const float*)d_in[5];
    const float* ln_b = (const float*)d_in[6];

    char* ws = (char*)d_ws;
    const size_t SZ_BF = (size_t)NTOK * DD * 2;   // 32 MB
    ushort_t* q    = (ushort_t*)(ws);
    ushort_t* k    = (ushort_t*)(ws + SZ_BF);
    ushort_t* v    = (ushort_t*)(ws + 2 * SZ_BF);
    float*    o    = (float*)(ws + 3 * SZ_BF);    // 64 MB fp32
    float*    beta = (float*)(ws + 3 * SZ_BF + (size_t)NTOK * DD * 4);

    // 1) projections (bf16 outputs; q pre-scaled, v SiLU'd)
    proj_gemm_kernel<<<2048, 256, 0, stream>>>(x, Wq, q, 0);
    proj_gemm_kernel<<<2048, 256, 0, stream>>>(x, Wk, k, 1);
    proj_gemm_kernel<<<2048, 256, 0, stream>>>(x, Wv, v, 2);
    // 2) beta = sigmoid(x @ Wb)
    beta_kernel<<<2048, 256, 0, stream>>>(x, Wb, beta);
    // 3) L2-normalize k rows
    knorm_kernel<<<2048, 256, 0, stream>>>(k);
    // 4) chunked delta-rule scan (8 batches x 32 value slices)
    scan_kernel<<<BB * NSLICE, 256, SMEM_SCAN, stream>>>(q, k, v, beta, o);
    // 5) LayerNorm -> d_out
    ln_kernel<<<2048, 256, 0, stream>>>(o, ln_g, ln_b, (float*)d_out);
}
